// AssignmentWeightedAverage_66159676227731
// MI455X (gfx1250) — 8-bit matrix operand order repaired and checked by reading, compiled, not run on hardware
//
#include <hip/hip_runtime.h>
#include <math.h>

#define N1 256
#define N2 256
#define IMG_H 480
#define IMG_W 854
#define HW (IMG_H * IMG_W)
#define DFEAT 512
#define KUNITS (HW / 64)
#define KSPLIT 105
#define UNITS_PER_BLOCK (KUNITS / KSPLIT)
#define FULL128_PER_BLOCK (UNITS_PER_BLOCK / 2)
static_assert(KSPLIT * UNITS_PER_BLOCK == KUNITS, "K split must be exact");
static_assert(HW % 64 == 0, "K must be multiple of 64");

#define INTER_SCALE 262144.0f

typedef int   v16i __attribute__((ext_vector_type(16)));
typedef float v8f  __attribute__((ext_vector_type(8)));

__global__ void k_zero_inter(float* __restrict__ inter) {
    int i = blockIdx.x * blockDim.x + threadIdx.x;
    if (i < N1 * N2) inter[i] = 0.0f;
}

__global__ __launch_bounds__(N2)
void k_box_iou(const float* __restrict__ b1, const float* __restrict__ b2,
               float* __restrict__ biou) {
    const int n = blockIdx.x;
    const int m = threadIdx.x;
    float x11 = b1[n * 4 + 0], y11 = b1[n * 4 + 1];
    float x12 = b1[n * 4 + 2], y12 = b1[n * 4 + 3];
    float x21 = b2[m * 4 + 0], y21 = b2[m * 4 + 1];
    float x22 = b2[m * 4 + 2], y22 = b2[m * 4 + 3];
    float a1 = (x12 - x11) * (y12 - y11);
    float a2 = (x22 - x21) * (y22 - y21);
    float lx = fmaxf(x11, x21), ly = fmaxf(y11, y21);
    float rx = fminf(x12, x22), ry = fminf(y12, y22);
    float w = fmaxf(rx - lx, 0.0f), h = fmaxf(ry - ly, 0.0f);
    float in = w * h;
    biou[n * N2 + m] = in / (a1 + a2 - in);
}

__global__ __launch_bounds__(256)
void k_mask_areas(const unsigned char* __restrict__ trk,
                  const unsigned char* __restrict__ cur,
                  int* __restrict__ area1, int* __restrict__ area2) {
    const unsigned char* base =
        (blockIdx.y == 0 ? trk : cur) + (size_t)blockIdx.x * HW;
    const uint4* p = reinterpret_cast<const uint4*>(base);
    const int n16 = HW / 16;
    int s = 0;
    for (int i = threadIdx.x; i < n16; i += blockDim.x) {
        uint4 v = p[i];
        s += __popc(v.x) + __popc(v.y) + __popc(v.z) + __popc(v.w);
    }
    __shared__ int red[256];
    red[threadIdx.x] = s;
    __syncthreads();
    for (int off = 128; off > 0; off >>= 1) {
        if (threadIdx.x < off) red[threadIdx.x] += red[threadIdx.x + off];
        __syncthreads();
    }
    if (threadIdx.x == 0) {
        (blockIdx.y == 0 ? area1 : area2)[blockIdx.x] = red[0];
    }
}

__device__ inline v16i load_a128(const unsigned char* p) {
    v16i f;
#pragma unroll
    for (int h = 0; h < 2; ++h) {
#pragma unroll
        for (int c = 0; c < 4; ++c) {
            uint2 q = *reinterpret_cast<const uint2*>(p + h * 64 + (c >> 1) * 32 + (c & 1) * 8);
            f[h * 8 + c * 2 + 0] = (int)q.x;
            f[h * 8 + c * 2 + 1] = (int)q.y;
        }
    }
    return f;
}

__device__ inline v16i load_a64_pad(const unsigned char* p) {
    v16i f;
#pragma unroll
    for (int c = 0; c < 4; ++c) {
        uint2 q = *reinterpret_cast<const uint2*>(p + (c >> 1) * 32 + (c & 1) * 8);
        f[c * 2 + 0] = (int)q.x;
        f[c * 2 + 1] = (int)q.y;
    }
#pragma unroll
    for (int j = 8; j < 16; ++j) f[j] = 0;
    return f;
}

__device__ inline v16i load_b128(const unsigned char* p) {
    v16i f;
#pragma unroll
    for (int c = 0; c < 4; ++c) {
        uint4 r = *reinterpret_cast<const uint4*>(p + c * 32);
        f[c * 4 + 0] = (int)r.x; f[c * 4 + 1] = (int)r.y;
        f[c * 4 + 2] = (int)r.z; f[c * 4 + 3] = (int)r.w;
    }
    return f;
}

__device__ inline v16i load_b64_pad(const unsigned char* p) {
    v16i f;
#pragma unroll
    for (int c = 0; c < 2; ++c) {
        uint4 r = *reinterpret_cast<const uint4*>(p + c * 32);
        f[c * 4 + 0] = (int)r.x; f[c * 4 + 1] = (int)r.y;
        f[c * 4 + 2] = (int)r.z; f[c * 4 + 3] = (int)r.w;
    }
#pragma unroll
    for (int j = 8; j < 16; ++j) f[j] = 0;
    return f;
}

__global__ __launch_bounds__(256)
void k_mask_inter_wmma(const unsigned char* __restrict__ trk,
                       const unsigned char* __restrict__ cur,
                       float* __restrict__ inter) {
    const int lane    = threadIdx.x & 31;
    const int wave    = threadIdx.x >> 5;
    const int wm      = wave & 1;
    const int wn      = wave >> 1;
    const int lane_lo = lane & 15;
    const int lane_hi = lane >> 4;

    const int m_base = blockIdx.x * 64 + wm * 32;
    const int n_base = wn * 64;

    const unsigned char* aptr[2];
#pragma unroll
    for (int t = 0; t < 2; ++t)
        aptr[t] = trk + (size_t)(m_base + t * 16 + lane_lo) * HW + lane_hi * 16;
    const unsigned char* bptr[4];
#pragma unroll
    for (int u = 0; u < 4; ++u)
        bptr[u] = cur + (size_t)(n_base + u * 16 + lane_lo) * HW + lane_hi * 16;

    v8f acc[2][4];
#pragma unroll
    for (int t = 0; t < 2; ++t)
#pragma unroll
        for (int u = 0; u < 4; ++u)
#pragma unroll
            for (int j = 0; j < 8; ++j) acc[t][u][j] = 0.0f;

    const size_t kbyte0 = (size_t)blockIdx.y * UNITS_PER_BLOCK * 64;

    for (int s = 0; s < FULL128_PER_BLOCK; ++s) {
        const size_t kb = kbyte0 + (size_t)s * 128;

        v16i a[2];
#pragma unroll
        for (int t = 0; t < 2; ++t) a[t] = load_a128(aptr[t] + kb);
        v16i b[4];
#pragma unroll
        for (int u = 0; u < 4; ++u) b[u] = load_b128(bptr[u] + kb);

#pragma unroll
        for (int t = 0; t < 2; ++t)
#pragma unroll
            for (int u = 0; u < 4; ++u)
                acc[t][u] = __builtin_amdgcn_wmma_f32_16x16x128_fp8_fp8(
                    a[t], b[u], (short)0, acc[t][u],
                     false,  false);
    }

    {
        const size_t kb = kbyte0 + (size_t)FULL128_PER_BLOCK * 128;
        v16i a[2];
#pragma unroll
        for (int t = 0; t < 2; ++t) a[t] = load_a64_pad(aptr[t] + kb);
        v16i b[4];
#pragma unroll
        for (int u = 0; u < 4; ++u) b[u] = load_b64_pad(bptr[u] + kb);
#pragma unroll
        for (int t = 0; t < 2; ++t)
#pragma unroll
            for (int u = 0; u < 4; ++u)
                acc[t][u] = __builtin_amdgcn_wmma_f32_16x16x128_fp8_fp8(
                    a[t], b[u], (short)0, acc[t][u], false, false);
    }

#pragma unroll
    for (int t = 0; t < 2; ++t) {
#pragma unroll
        for (int u = 0; u < 4; ++u) {
            const int R = m_base + t * 16 + lane_hi * 8;
            const int C = n_base + u * 16 + lane_lo;
#pragma unroll
            for (int j = 0; j < 8; ++j)
                atomicAdd(&inter[(R + j) * N2 + C], acc[t][u][j]);
        }
    }
}

__global__ __launch_bounds__(N2)
void k_combine(const float* __restrict__ f1, const float* __restrict__ f2,
               const float* __restrict__ biou, const float* __restrict__ inter,
               const int* __restrict__ area1, const int* __restrict__ area2,
               float* __restrict__ out) {
    const int n = blockIdx.x;
    const int m = threadIdx.x;

    const float bi = biou[n * N2 + m];
    const float in_f =
        (bi > 0.0f) ? inter[n * N2 + m] * INTER_SCALE : 0.0f;
    const float un = (float)area1[n] + (float)area2[m] - in_f;
    const float mi = in_f / un;

    const float4* p1 = reinterpret_cast<const float4*>(f1 + (size_t)n * DFEAT);
    const float4* p2 = reinterpret_cast<const float4*>(f2 + (size_t)m * DFEAT);
    float d2 = 0.0f;
#pragma unroll 4
    for (int i = 0; i < DFEAT / 4; ++i) {
        float4 a = p1[i], b = p2[i];
        float dx = a.x - b.x, dy = a.y - b.y, dz = a.z - b.z, dw = a.w - b.w;
        d2 += dx * dx + dy * dy + dz * dz + dw * dw;
    }
    const float dist = sqrtf(fmaxf(d2, 0.0f));

    out[n * N2 + m] = 0.4f * (1.0f - bi) + 0.3f * (1.0f - mi) + 0.3f * dist;
}

extern "C" void kernel_launch(void* const* d_in, const int* in_sizes, int n_in,
                              void* d_out, int out_size, void* d_ws, size_t ws_size,
                              hipStream_t stream) {
    const float* trk_f = (const float*)d_in[0];
    const float* cur_f = (const float*)d_in[1];
    const float* trk_b = (const float*)d_in[2];
    const float* cur_b = (const float*)d_in[3];
    const unsigned char* trk_m = (const unsigned char*)d_in[6];
    const unsigned char* cur_m = (const unsigned char*)d_in[7];
    float* out = (float*)d_out;

    char* ws = (char*)d_ws;
    float* inter = (float*)ws;
    float* biou  = (float*)(ws + (size_t)N1 * N2 * 4);
    int*   area1 = (int*)(ws + (size_t)2 * N1 * N2 * 4);
    int*   area2 = area1 + N1;

    k_zero_inter<<<dim3((N1 * N2 + 255) / 256), dim3(256), 0, stream>>>(inter);
    k_box_iou<<<dim3(N1), dim3(N2), 0, stream>>>(trk_b, cur_b, biou);
    k_mask_areas<<<dim3(N1, 2), dim3(256), 0, stream>>>(trk_m, cur_m, area1, area2);
    k_mask_inter_wmma<<<dim3(N1 / 64, KSPLIT), dim3(256), 0, stream>>>(trk_m, cur_m, inter);
    k_combine<<<dim3(N1), dim3(N2), 0, stream>>>(trk_f, cur_f, biou, inter,
                                                 area1, area2, out);
}
